// Transformer_51049981280446
// MI455X (gfx1250) — matrix operand order repaired and checked by reading, not compiled, not run on hardware
//
#include <hip/hip_runtime.h>
#include <hip/hip_bf16.h>

typedef __attribute__((ext_vector_type(16))) _Float16 v16h;
typedef __attribute__((ext_vector_type(8)))  _Float16 v8h;
typedef __attribute__((ext_vector_type(4)))  _Float16 v4h;
typedef __attribute__((ext_vector_type(8)))  float    v8f;
typedef __attribute__((ext_vector_type(4)))  float    v4f;

#define EMB 1024
#define HEADS 16
#define HD 64
#define BB 4
#define SQL 2048
#define SKL 2048
#define MTOT (BB * SQL)   // 8192 total rows

__device__ __forceinline__ v8f wmma_f16(v16h a, v16h b, v8f c) {
    return __builtin_amdgcn_wmma_f32_16x16x32_f16(
        false, a, false, b, (short)0, c, false, false);
}

// A-fragment 16x32 f16 (ISA 7.12.2): per lane two contiguous 8-half chunks at
// k = 16*hg and k = 16*hg + 8 (this file's contiguous order k = 16*hg + s, same as the B loads).
__device__ __forceinline__ v16h mk_a_f16(v8h lo, v8h hi) {
    v16h a;
    #pragma unroll
    for (int e = 0; e < 8; ++e) { a[e] = lo[e]; a[8 + e] = hi[e]; }
    return a;
}

// ---------------------------------------------------------------------------
// Elementwise pre-passes (memory bound; ~7us total at HBM peak)
// ---------------------------------------------------------------------------
__global__ __launch_bounds__(256) void cvt_kernel(
    const float* __restrict__ s, _Float16* __restrict__ d, int n4)
{
    int i = blockIdx.x * 256 + threadIdx.x;
    if (i < n4) {
        v4f x = ((const v4f*)s)[i];
        v4h y;
        #pragma unroll
        for (int e = 0; e < 4; ++e) y[e] = (_Float16)x[e];
        ((v4h*)d)[i] = y;
    }
}

// bias = -log(dist + 1e-9) * |scale|, f16 (shared across all 16 heads)
__global__ __launch_bounds__(256) void bias_kernel(
    const float* __restrict__ dist, const float* __restrict__ dsc,
    _Float16* __restrict__ bh, int n4)
{
    int i = blockIdx.x * 256 + threadIdx.x;
    if (i < n4) {
        float sabs = fabsf(dsc[0]);
        v4f x = ((const v4f*)dist)[i];
        v4h y;
        #pragma unroll
        for (int e = 0; e < 4; ++e)
            y[e] = (_Float16)(-__logf(x[e] + 1e-9f) * sabs);
        ((v4h*)bh)[i] = y;
    }
}

// ---------------------------------------------------------------------------
// Unified GEMM: Y = (Xh @ Wh^T + bias) * scale, Xh/Wh f16, fp32 accumulate.
// One wave -> 16(M) x 64(N); pure b128 load + WMMA inner loop (no conversions).
// ---------------------------------------------------------------------------
__global__ __launch_bounds__(256) void gemm_kernel(
    const _Float16* __restrict__ Xh, const _Float16* __restrict__ Wh,
    const float* __restrict__ bias,
    _Float16* __restrict__ Yrow, _Float16* __restrict__ Ycol,
    float* __restrict__ Yf32, float scale)
{
    const int lane = threadIdx.x & 31;
    const int wave = threadIdx.x >> 5;
    const int gw   = blockIdx.x * 8 + wave;      // 8192 waves
    const int m0 = (gw >> 4) * 16;               // 512 M-tiles
    const int n0 = (gw & 15) * 64;               // 16 N-groups of 64
    const int Ml = lane & 15;
    const int hg = lane >> 4;

    v8f acc[4] = {};

    #pragma unroll 2
    for (int kk = 0; kk < EMB; kk += 32) {
        const _Float16* xr = Xh + (size_t)(m0 + Ml) * EMB + kk;
        v8h a0 = *(const v8h*)(xr + 16 * hg);
        v8h a1 = *(const v8h*)(xr + 8 + 16 * hg);
        v16h wf[4];
        #pragma unroll
        for (int t = 0; t < 4; ++t)
            wf[t] = *(const v16h*)(Wh + (size_t)(n0 + t * 16 + Ml) * EMB + kk + 16 * hg);
        v16h a = mk_a_f16(a0, a1);
        #pragma unroll
        for (int t = 0; t < 4; ++t)
            acc[t] = wmma_f16(a, wf[t], acc[t]);
    }

    #pragma unroll
    for (int t = 0; t < 4; ++t) {
        int n = n0 + t * 16 + Ml;
        float bv = bias[n];
        if (Ycol) {                     // transposed store: contiguous 8 halfs
            v8h col;
            #pragma unroll
            for (int j = 0; j < 8; ++j)
                col[j] = (_Float16)((acc[t][j] + bv) * scale);
            *(v8h*)(Ycol + (size_t)n * MTOT + m0 + 8 * hg) = col;
        }
        #pragma unroll
        for (int j = 0; j < 8; ++j) {
            float r = (acc[t][j] + bv) * scale;
            int row = m0 + 8 * hg + j;
            if (Yrow) Yrow[(size_t)row * EMB + n] = (_Float16)r;
            if (Yf32) Yf32[(size_t)row * EMB + n] = r;
        }
    }
}

// ---------------------------------------------------------------------------
// Flash attention: one wave per (b, h, 16 q-rows); K in steps of 32.
// Precomputed f16 bias tile staged via global_load_async_to_lds_b128.
// ---------------------------------------------------------------------------
#define PSTR 40   // LDS stride in halfs (80B: 16B chunk alignment, bank-split)

__global__ __launch_bounds__(256) void attn_kernel(
    const _Float16* __restrict__ Qh, const _Float16* __restrict__ Kh,
    const _Float16* __restrict__ Vt, const _Float16* __restrict__ biasH,
    _Float16* __restrict__ Oh)
{
    __shared__ __align__(16) _Float16 pl[8][16 * PSTR];   // per-wave P tile
    __shared__ __align__(16) _Float16 bt[8][16 * PSTR];   // per-wave bias tile

    const int lane = threadIdx.x & 31;
    const int wave = threadIdx.x >> 5;
    const int gw   = blockIdx.x * 8 + wave;   // 8192 waves
    const int b  = gw >> 11;
    const int h  = (gw >> 7) & 15;
    const int q0 = (gw & 127) * 16;
    const int Ml = lane & 15;
    const int hg = lane >> 4;

    // Q A-fragments for d=0..31 and 32..63
    const _Float16* qr = Qh + (size_t)(b * SQL + q0 + Ml) * EMB + h * HD;
    v16h aq0 = mk_a_f16(*(const v8h*)(qr +      16 * hg), *(const v8h*)(qr + 8 + 16 * hg));
    v16h aq1 = mk_a_f16(*(const v8h*)(qr + 32 + 16 * hg), *(const v8h*)(qr + 40 + 16 * hg));

    float mrow[8], lrow[8];
    #pragma unroll
    for (int j = 0; j < 8; ++j) { mrow[j] = -1e30f; lrow[j] = 0.0f; }
    v8f o[4] = {};

    _Float16* plw = &pl[wave][0];
    _Float16* btw = &bt[wave][0];
    const uint32_t lds_b = (uint32_t)(uintptr_t)btw;

    // coalesced async staging map: lane -> (row = lane>>2 + 8*issue, 8 halfs
    // at col (lane&3)*8); 2 issues cover the 16x32 f16 bias tile (1KB).
    const int brow = lane >> 2;
    const int bcol = (lane & 3) * 8;

    for (int kc = 0; kc < SKL; kc += 32) {
        // ---- async-stage bias tile into LDS (2 x b128 per lane)
        #pragma unroll
        for (int it = 0; it < 2; ++it) {
            uint64_t ga = (uint64_t)(uintptr_t)
                (biasH + ((size_t)b * SQL + q0 + brow + 8 * it) * SKL + kc + bcol);
            uint32_t la = lds_b + (uint32_t)(((brow + 8 * it) * PSTR + bcol) * 2);
            asm volatile("global_load_async_to_lds_b128 %0, %1, off"
                         :: "v"(la), "v"(ga) : "memory");
        }

        // ---- scores S = Q.K^T : two 16x16 tiles
        const _Float16* kr0 = Kh + (size_t)(b * SKL + kc + Ml) * EMB + h * HD;
        const _Float16* kr1 = kr0 + (size_t)16 * EMB;
        v16h b00 = *(const v16h*)(kr0 + 16 * hg);
        v16h b01 = *(const v16h*)(kr0 + 32 + 16 * hg);
        v16h b10 = *(const v16h*)(kr1 + 16 * hg);
        v16h b11 = *(const v16h*)(kr1 + 32 + 16 * hg);
        __builtin_prefetch(kr0 + (size_t)32 * EMB, 0, 0);
        v8f s0 = {}, s1 = {};
        s0 = wmma_f16(aq0, b00, s0);
        s0 = wmma_f16(aq1, b01, s0);
        s1 = wmma_f16(aq0, b10, s1);
        s1 = wmma_f16(aq1, b11, s1);

        // ---- add precomputed bias from LDS
        asm volatile("s_wait_asynccnt 0" ::: "memory");
        #pragma unroll
        for (int j = 0; j < 8; ++j) {
            int r = 8 * hg + j;
            s0[j] += (float)btw[r * PSTR + Ml];
            s1[j] += (float)btw[r * PSTR + 16 + Ml];
        }

        // ---- online softmax per row (row = 8*hg + j; N over 16 lanes)
        #pragma unroll
        for (int j = 0; j < 8; ++j) {
            float mt = fmaxf(s0[j], s1[j]);
            #pragma unroll
            for (int off = 8; off > 0; off >>= 1)
                mt = fmaxf(mt, __shfl_xor(mt, off, 16));
            float mnew = fmaxf(mrow[j], mt);
            float corr = __expf(mrow[j] - mnew);
            float p0 = __expf(s0[j] - mnew);
            float p1 = __expf(s1[j] - mnew);
            float ps = p0 + p1;
            #pragma unroll
            for (int off = 8; off > 0; off >>= 1)
                ps += __shfl_xor(ps, off, 16);
            lrow[j] = lrow[j] * corr + ps;
            mrow[j] = mnew;
            #pragma unroll
            for (int t = 0; t < 4; ++t) o[t][j] *= corr;
            int pr = 8 * hg + j;
            plw[pr * PSTR + Ml]      = (_Float16)p0;
            plw[pr * PSTR + 16 + Ml] = (_Float16)p1;
        }

        // ---- reload P as 16x32 A fragment (same-wave LDS is in-order)
        v16h ap = mk_a_f16(*(const v8h*)(plw + Ml * PSTR +     16 * hg),
                           *(const v8h*)(plw + Ml * PSTR + 8 + 16 * hg));

        // ---- O += P @ V with transposed V (contiguous 32B per lane)
        #pragma unroll
        for (int t = 0; t < 4; ++t) {
            const _Float16* vr = Vt + (size_t)(h * HD + t * 16 + Ml) * MTOT
                                    + (size_t)b * SKL + kc + 16 * hg;
            v16h bv = *(const v16h*)vr;
            __builtin_prefetch(vr + 32, 0, 0);
            o[t] = wmma_f16(ap, bv, o[t]);
        }
    }

    // ---- epilogue: normalize and store attn output (f16 row-major)
    #pragma unroll
    for (int t = 0; t < 4; ++t) {
        #pragma unroll
        for (int j = 0; j < 8; ++j) {
            float ov = o[t][j] / lrow[j];
            Oh[(size_t)(b * SQL + q0 + 8 * hg + j) * EMB + h * HD + t * 16 + Ml] =
                (_Float16)ov;
        }
    }
}

// ---------------------------------------------------------------------------
extern "C" void kernel_launch(void* const* d_in, const int* in_sizes, int n_in,
                              void* d_out, int out_size, void* d_ws, size_t ws_size,
                              hipStream_t stream) {
    (void)in_sizes; (void)n_in; (void)out_size; (void)ws_size;

    const float* query = (const float*)d_in[0];
    const float* key_  = (const float*)d_in[1];
    const float* value = (const float*)d_in[2];
    const float* dist  = (const float*)d_in[3];
    const float* Wq    = (const float*)d_in[4];
    const float* bq    = (const float*)d_in[5];
    const float* Wk    = (const float*)d_in[6];
    const float* bk    = (const float*)d_in[7];
    const float* Wv    = (const float*)d_in[8];
    const float* bv    = (const float*)d_in[9];
    const float* Wo    = (const float*)d_in[10];
    const float* bo    = (const float*)d_in[11];
    const float* dsc   = (const float*)d_in[12];
    float* out = (float*)d_out;

    // workspace carve (f16): Qh, Kh, Vt, Oh (Oh doubles as the input-activation
    // staging buffer Xh during the projection phase), biasH, Wh (reused 4x).
    const size_t ME = (size_t)MTOT * EMB;          // 8M elements
    _Float16* Qh    = (_Float16*)d_ws;
    _Float16* Kh    = Qh + ME;
    _Float16* Vt    = Kh + ME;
    _Float16* Oh    = Vt + ME;                     // alias: Xh during proj
    _Float16* biasH = Oh + ME;                     // 16.7M elements
    _Float16* Wh    = biasH + (size_t)BB * SQL * SKL;  // 1M elements
    _Float16* Xh    = Oh;

    dim3 blk(256);
    dim3 grd(1024);            // 8192 waves per GEMM/attention phase
    const int nX4 = (int)(ME / 4);                 // 2M vec4 per activation
    const int nW4 = (int)((size_t)EMB * EMB / 4);  // 256K vec4 per weight
    const int nD4 = (int)((size_t)BB * SQL * SKL / 4);
    dim3 gX((nX4 + 255) / 256), gW((nW4 + 255) / 256), gD((nD4 + 255) / 256);

    _Float16* nullh = (_Float16*)nullptr;
    float*    nullf = (float*)nullptr;

    // Q projection (scaled by head_dim^-0.5 = 1/8)
    cvt_kernel<<<gX, blk, 0, stream>>>(query, Xh, nX4);
    cvt_kernel<<<gW, blk, 0, stream>>>(Wq, Wh, nW4);
    gemm_kernel<<<grd, blk, 0, stream>>>(Xh, Wh, bq, Qh, nullh, nullf, 0.125f);
    // K projection
    cvt_kernel<<<gX, blk, 0, stream>>>(key_, Xh, nX4);
    cvt_kernel<<<gW, blk, 0, stream>>>(Wk, Wh, nW4);
    gemm_kernel<<<grd, blk, 0, stream>>>(Xh, Wh, bk, Kh, nullh, nullf, 1.0f);
    // V projection, stored transposed
    cvt_kernel<<<gX, blk, 0, stream>>>(value, Xh, nX4);
    cvt_kernel<<<gW, blk, 0, stream>>>(Wv, Wh, nW4);
    gemm_kernel<<<grd, blk, 0, stream>>>(Xh, Wh, bv, nullh, Vt, nullf, 1.0f);

    // distance bias pre-pass (shared across heads)
    bias_kernel<<<gD, blk, 0, stream>>>(dist, dsc, biasH, nD4);

    // fused scores + bias + online softmax + PV
    attn_kernel<<<grd, blk, 0, stream>>>(Qh, Kh, Vt, biasH, Oh);

    // output projection to fp32
    cvt_kernel<<<gW, blk, 0, stream>>>(Wo, Wh, nW4);
    gemm_kernel<<<grd, blk, 0, stream>>>(Oh, Wh, bo, nullh, nullh, out, 1.0f);
}
